// EntropyEncoder_84928683311459
// MI455X (gfx1250) — 8-bit matrix operand order repaired and checked by reading, compiled, not run on hardware
//
#include <hip/hip_runtime.h>
#include <stdint.h>

typedef __attribute__((ext_vector_type(8))) int v8i;

#define N_BITS_TOT   (1u << 20)
#define NBLK         1024
#define CHUNK        1024
#define NKEYS        512
#define ATOTAL       4096ull
#define MAXBITS      (1u << 21)
#define PACKED_BYTES (MAXBITS / 8)
#define FULL_R       (1ull << 32)
#define HALF_R       (1ull << 31)
#define QUART_R      (1ull << 30)
#define THREEQ_R     (3ull << 30)
#define RMASK        (FULL_R - 1ull)

__global__ void k_keys(const float* __restrict__ sym, uint16_t* __restrict__ keys, int n) {
    int i = blockIdx.x * blockDim.x + threadIdx.x;
    if (i >= n) return;
    uint32_t bit = (sym[i] > 0.0f) ? 1u : 0u;
    uint32_t ctx = 0u;
#pragma unroll
    for (int k = 1; k <= 8; ++k) {
        uint32_t pb = (i >= k) ? ((sym[i - k] > 0.0f) ? 1u : 0u) : 0u;
        ctx |= pb << (k - 1);
    }
    keys[i] = (uint16_t)((ctx << 1) | bit);
}

__global__ void __launch_bounds__(32) k_hist(const uint16_t* __restrict__ keys,
                                             uint32_t* __restrict__ hist) {
    __shared__ uint16_t lk[CHUNK];
    const int l   = threadIdx.x;
    const int blk = blockIdx.x;
    const uint16_t* src = keys + blk * CHUNK;
    for (int j = l; j < CHUNK; j += 32) lk[j] = src[j];
    __syncthreads();

    v8i cLo = {}; v8i cHi = {};
    const int M      = l & 15;
    const int N      = l & 15;
    const int laneHi = (l >> 4) & 1;

    for (int t = 0; t < CHUNK / 64; ++t) {
        const uint16_t* kt = &lk[t * 64];
        v8i aLo, aHi, bM;
#pragma unroll
        for (int v = 0; v < 8; ++v) {
            int abase = ((v >= 4) ? 32 : 0) + ((v & 1) * 4) + (((v >> 1) & 1) * 16)
                        + (laneHi ? 8 : 0);
            uint32_t wLo = 0u, wHi = 0u;
#pragma unroll
            for (int byte = 0; byte < 4; ++byte) {
                uint32_t kv = kt[abase + byte];
                uint32_t g  = (((kv >> 4) & 0xFu) == (uint32_t)M) ? 1u : 0u;
                wLo |= (g & (((kv >> 8) == 0u) ? 1u : 0u)) << (8 * byte);
                wHi |= (g & (((kv >> 8) == 1u) ? 1u : 0u)) << (8 * byte);
            }
            aLo[v] = (int)wLo;
            aHi[v] = (int)wHi;
            int bbase = ((v >= 4) ? 32 : 0) + ((v & 1) * 4) + (((v >> 1) & 1) * 16) + (laneHi ? 8 : 0);
            uint32_t wB = 0u;
#pragma unroll
            for (int byte = 0; byte < 4; ++byte) {
                uint32_t kv = kt[bbase + byte];
                wB |= ((((kv & 0xFu) == (uint32_t)N) ? 1u : 0u)) << (8 * byte);
            }
            bM[v] = (int)wB;
        }
        cLo = __builtin_amdgcn_wmma_i32_16x16x64_iu8(false, aLo, false, bM, cLo, false, false);
        cHi = __builtin_amdgcn_wmma_i32_16x16x64_iu8(false, aHi, false, bM, cHi, false, false);
    }

    uint32_t* dst = hist + blk * NKEYS;
#pragma unroll
    for (int r = 0; r < 8; ++r) {
        int Mm = r + (laneHi ? 8 : 0);
        dst[0 * 256 + Mm * 16 + N] = (uint32_t)cLo[r];
        dst[256     + Mm * 16 + N] = (uint32_t)cHi[r];
    }
}

__global__ void k_scan(const uint32_t* __restrict__ hist, uint32_t* __restrict__ off) {
    int k = blockIdx.x * blockDim.x + threadIdx.x;
    if (k >= NKEYS) return;
    uint32_t run = 0u;
    for (int b = 0; b < NBLK; ++b) {
        off[b * NKEYS + k] = run;
        run += hist[b * NKEYS + k];
    }
}

__global__ void __launch_bounds__(32) k_freqs(const uint16_t* __restrict__ keys,
                                              const uint32_t* __restrict__ off,
                                              uint16_t* __restrict__ freqs) {
    __shared__ uint32_t cnt[NKEYS];
    __shared__ uint16_t lk[CHUNK];
    const int l = threadIdx.x;
    const int blk = blockIdx.x;
    for (int j = l; j < NKEYS; j += 32) cnt[j] = off[blk * NKEYS + j];

    {
        uint32_t lds_base = (uint32_t)(uintptr_t)(&lk[0]);
        const uint8_t* gsrc = (const uint8_t*)(keys + blk * CHUNK);
#pragma unroll
        for (int t = 0; t < 8; ++t) {
            uint32_t ldsa = lds_base + (uint32_t)(l * 64 + t * 8);
            uint64_t ga   = (uint64_t)(uintptr_t)(gsrc + l * 64 + t * 8);
            asm volatile("global_load_async_to_lds_b64 %0, %1, off"
                         :: "v"(ldsa), "v"(ga) : "memory");
        }
        asm volatile("s_wait_asynccnt 0" ::: "memory");
    }
    __syncthreads();

    if (l == 0) {
        uint16_t* fdst = freqs + blk * CHUNK;
        for (int i = 0; i < CHUNK; ++i) {
            uint32_t k = lk[i];
            uint32_t a = 1u + cnt[k & ~1u];
            uint32_t b = 1u + cnt[k |  1u];
            double p0 = (double)a / (double)(a + b);
            double f  = rint(p0 * (double)(ATOTAL - 2ull)) + 1.0;
            if (f < 1.0) f = 1.0;
            if (f > (double)(ATOTAL - 1ull)) f = (double)(ATOTAL - 1ull);
            fdst[i] = (uint16_t)f;
            cnt[k] += 1u;
        }
    }
}

__global__ void k_zero(uint64_t* __restrict__ p, int n64) {
    int i = blockIdx.x * blockDim.x + threadIdx.x;
    if (i < n64) p[i] = 0ull;
}

__global__ void k_encode(const uint16_t* __restrict__ keys,
                         const uint16_t* __restrict__ freqs,
                         uint8_t* __restrict__ buf,
                         unsigned long long* __restrict__ nbits) {
    if (threadIdx.x != 0 || blockIdx.x != 0) return;
    uint64_t low = 0ull, high = FULL_R - 1ull;
    uint64_t pending = 0ull, ptr = 0ull;
    for (uint32_t i = 0; i < N_BITS_TOT; ++i) {
        if (i + 256 < N_BITS_TOT) {
            __builtin_prefetch(&freqs[i + 256], 0, 0);
            __builtin_prefetch(&keys[i + 256], 0, 0);
        }
        uint32_t bit = keys[i] & 1u;
        uint64_t p0  = (uint64_t)freqs[i];
        uint64_t rng   = high - low + 1ull;
        uint64_t split = low + ((rng * p0) >> 12);
        if (bit == 0u) high = split - 1ull; else low = split;
        for (;;) {
            bool is_a = (high < HALF_R);
            bool is_b = (!is_a) && (low >= HALF_R);
            bool is_c = (!is_a) && (!is_b) && (low >= QUART_R) && (high < THREEQ_R);
            if (!(is_a || is_b || is_c)) break;
            if (is_a || is_b) {
                uint8_t ob = is_b ? 1u : 0u;
                buf[ptr++] = ob;
                for (uint64_t j = 0; j < pending; ++j) buf[ptr++] = (uint8_t)(1u - ob);
                pending = 0ull;
            } else {
                pending += 1ull;
            }
            uint64_t shift = is_a ? 0ull : (is_b ? HALF_R : QUART_R);
            low  = ((low  - shift) << 1) & RMASK;
            high = (((high - shift) << 1) & RMASK) | 1ull;
        }
    }
    pending += 1ull;
    uint8_t fb = (low >= QUART_R) ? 1u : 0u;
    buf[ptr] = fb;
    for (uint64_t j = 0; j < pending; ++j) buf[ptr + 1 + j] = (uint8_t)(1u - fb);
    ptr += 1ull + pending;
    *nbits = ptr;
}

__global__ void k_pack(const uint8_t* __restrict__ buf,
                       const unsigned long long* __restrict__ nbits,
                       float* __restrict__ out) {
    int i = blockIdx.x * blockDim.x + threadIdx.x;
    if (i < (int)PACKED_BYTES) {
        uint32_t v = 0u;
#pragma unroll
        for (int j = 0; j < 8; ++j) v |= ((uint32_t)buf[i * 8 + j]) << (7 - j);
        out[i] = (float)v;
    }
    if (i == 0) out[PACKED_BYTES] = (float)(*nbits);
}

extern "C" void kernel_launch(void* const* d_in, const int* in_sizes, int n_in,
                              void* d_out, int out_size, void* d_ws, size_t ws_size,
                              hipStream_t stream) {
    const float* sym = (const float*)d_in[0];
    float* out = (float*)d_out;
    (void)in_sizes; (void)n_in; (void)out_size; (void)ws_size;

    uint8_t* ws = (uint8_t*)d_ws;
    uint16_t* keys  = (uint16_t*)(ws + 0);
    uint16_t* freqs = (uint16_t*)(ws + (2u << 20));
    uint32_t* hist  = (uint32_t*)(ws + (4u << 20));
    uint32_t* off   = (uint32_t*)(ws + (6u << 20));
    uint8_t*  bitbf = (uint8_t*) (ws + (8u << 20));
    unsigned long long* nbits = (unsigned long long*)(ws + (10u << 20));

    k_keys<<<(N_BITS_TOT + 255) / 256, 256, 0, stream>>>(sym, keys, (int)N_BITS_TOT);
    k_hist<<<NBLK, 32, 0, stream>>>(keys, hist);
    k_scan<<<(NKEYS + 255) / 256, 256, 0, stream>>>(hist, off);
    k_freqs<<<NBLK, 32, 0, stream>>>(keys, off, freqs);
    k_zero<<<(int)((MAXBITS / 8 + 255) / 256), 256, 0, stream>>>((uint64_t*)bitbf,
                                                                 (int)(MAXBITS / 8));
    k_encode<<<1, 1, 0, stream>>>(keys, freqs, bitbf, nbits);
    k_pack<<<(PACKED_BYTES + 255) / 256, 256, 0, stream>>>(bitbf, nbits, out);
}
